// MLP_47098611368461
// MI455X (gfx1250) — matrix operand order repaired and checked by reading, not compiled, not run on hardware
//
#include <hip/hip_runtime.h>
#include <math.h>

typedef __attribute__((ext_vector_type(16))) _Float16 v16h;
typedef __attribute__((ext_vector_type(8)))  _Float16 v8h;
typedef __attribute__((ext_vector_type(8)))  float    v8f;

union V16U { v16h v; v8h h[2]; };

#define DIMC  64
#define HIDC  256
#define IMGH  128
#define IMGW  128
#define TW    16           // interior tile width
#define THh   8            // interior tile height
#define HLW   18           // halo width  (TW + 2)
#define HLH   10           // halo height (THh + 2)
#define NPIX  180          // HLW * HLH
#define NPIXP 192          // padded to multiple of 16 (12 N-tiles)
#define XNS   72           // xn row stride (halfs), 144B: 16B-aligned, conflict-free
#define W1S   72           // fc1_w row stride (halfs)
#define CHS   264          // 256-channel row stride (halfs), 528B

// Branchless erf-GELU: Abramowitz-Stegun 7.1.26 (|err| < 1.5e-7, far below the
// f16 operand rounding of the WMMA GEMMs). 1x v_rcp_f32 + 1x v_exp_f32 (TRANS,
// co-issues with VALU) + ~12 FMAs; no exec-mask divergence in the hot loops.
__device__ __forceinline__ float gelu_exact(float v) {
  const float z  = v * 0.70710678118654752440f;
  const float az = fabsf(z);
  const float t  = __builtin_amdgcn_rcpf(fmaf(0.3275911f, az, 1.0f));
  float p = fmaf(1.061405429f, t, -1.453152027f);
  p = fmaf(p, t, 1.421413741f);
  p = fmaf(p, t, -0.284496736f);
  p = fmaf(p, t, 0.254829592f);
  p = p * t;
  const float e  = __builtin_amdgcn_exp2f(az * az * -1.4426950408889634f);
  const float ea = fmaf(-p, e, 1.0f);          // erf(|z|)
  const float er = copysignf(ea, z);           // erf(z)
  return 0.5f * v * (1.0f + er);
}

__global__ __launch_bounds__(256, 1)
void convnext_fused(const float* __restrict__ x,
                    const float* __restrict__ ln_w, const float* __restrict__ ln_b,
                    const float* __restrict__ fc1_w, const float* __restrict__ fc1_b,
                    const float* __restrict__ pos_w, const float* __restrict__ pos_b,
                    const float* __restrict__ fc2_w, const float* __restrict__ fc2_b,
                    const float* __restrict__ s_sx, const float* __restrict__ b_sx,
                    const float* __restrict__ s_sy, const float* __restrict__ b_sy,
                    const float* __restrict__ s_lp, const float* __restrict__ b_lp,
                    float* __restrict__ y)
{
  __shared__ __align__(16) _Float16 sXn[NPIXP * XNS];   // xn (B-operand of fc1), [pixel][ch]
  __shared__ __align__(16) _Float16 sH [NPIX  * CHS];   // h after gelu, [pixel][ch]
  __shared__ __align__(16) _Float16 sW1[HIDC * W1S];    // fc1_w f16, [m][k]
  __shared__ __align__(16) _Float16 sW2[DIMC * CHS];    // fc2_w f16, [m][k]
  __shared__ __align__(16) _Float16 sStage[32 * CHS];   // h' for 32 pixels (fc2 B-operand)
  __shared__ float sWt[9 * HIDC];                       // combined dw conv weights, [tap][ch]
  __shared__ float sBt[HIDC];
  __shared__ float sB1[HIDC];
  __shared__ float sB2[DIMC];
  __shared__ float sLnW[DIMC];
  __shared__ float sLnB[DIMC];
  __shared__ float sValid[NPIXP];

  const int tid  = threadIdx.x;
  const int wave = tid >> 5;
  const int lane = tid & 31;
  const int lq   = lane & 15;   // N/M index within tile
  const int lh   = lane >> 4;   // half-wave select
  const int bb   = blockIdx.z;
  const int tX   = blockIdx.x;
  const int tY   = blockIdx.y;

  // ---------- Phase 0: stage weights into LDS (f16 for WMMA, f32 for conv) ----------
  for (int i = tid; i < HIDC * DIMC; i += 256) {
    int m = i >> 6, k = i & 63;
    sW1[m * W1S + k] = (_Float16)fc1_w[i];
  }
  for (int i = tid; i < DIMC * HIDC; i += 256) {
    int m = i >> 8, k = i & 255;
    sW2[m * CHS + k] = (_Float16)fc2_w[i];
  }
  if (tid < HIDC) {
    const int ch = tid;
    const float MSX[9] = {0.f, 1.f, -1.f,  2.f, 0.f, -2.f,  1.f, 0.f, -1.f};
    const float MSY[9] = {1.f, 2.f,  1.f,  0.f, 0.f,  0.f, -1.f,-2.f, -1.f};
    const float MLP[9] = {1.f, 0.f,  0.f,  1.f,-4.f,  1.f,  0.f, 1.f,  0.f};
    const float sx = s_sx[ch], sy = s_sy[ch], sl = s_lp[ch];
#pragma unroll
    for (int t = 0; t < 9; ++t)
      sWt[t * HIDC + ch] = pos_w[ch * 9 + t] + sx * MSX[t] + sy * MSY[t] + sl * MLP[t];
    sBt[ch] = pos_b[ch] + b_sx[ch] + b_sy[ch] + b_lp[ch];
    sB1[ch] = fc1_b[ch];
  }
  if (tid < DIMC) { sB2[tid] = fc2_b[tid]; sLnW[tid] = ln_w[tid]; sLnB[tid] = ln_b[tid]; }
  __syncthreads();

  // ---------- Phase 1: per-pixel LayerNorm over C=64, store f16 xn ----------
  if (tid < NPIXP) {
    const int p = tid;
    bool inImg = false; int gy = 0, gx = 0;
    if (p < NPIX) {
      const int hy = p / HLW, hx = p - hy * HLW;
      gy = tY * THh - 1 + hy;
      gx = tX * TW  - 1 + hx;
      inImg = (gy >= 0) && (gy < IMGH) && (gx >= 0) && (gx < IMGW);
    }
    sValid[p] = inImg ? 1.0f : 0.0f;
    if (inImg) {
      const float* xb = x + (((size_t)bb * DIMC) * IMGH + gy) * IMGW + gx;
      float sum = 0.f, sq = 0.f;
#pragma unroll 8
      for (int c = 0; c < DIMC; ++c) {
        float v = xb[(size_t)c * (IMGH * IMGW)];
        sum += v; sq += v * v;
      }
      const float mean = sum * (1.0f / DIMC);
      const float var  = sq  * (1.0f / DIMC) - mean * mean;
      const float rstd = rsqrtf(var + 1e-6f);
      for (int c0 = 0; c0 < DIMC; c0 += 8) {
        v8h o;
#pragma unroll
        for (int j = 0; j < 8; ++j) {
          float v = xb[(size_t)(c0 + j) * (IMGH * IMGW)];
          o[j] = (_Float16)(sLnW[c0 + j] * ((v - mean) * rstd) + sLnB[c0 + j]);
        }
        *(v8h*)&sXn[p * XNS + c0] = o;
      }
    } else {
      v8h z = {};
      for (int c0 = 0; c0 < DIMC; c0 += 8) *(v8h*)&sXn[p * XNS + c0] = z;
    }
  }
  __syncthreads();

  // ---------- Phase 2: fc1 via WMMA f16 (H[256 x 192] = W1[256x64] * Xn[64x192]) ----------
  // 16 M-tiles x 12 N-tiles, distributed over 8 waves; K = 64 = 2 steps of 32.
  for (int t = wave; t < 16 * (NPIXP / 16); t += 8) {
    const int mT = t & 15;
    const int nT = t >> 4;
    const int row = mT * 16 + lq;        // output channel row for A
    const int col = nT * 16 + lq;        // pixel column for B
    v8f acc = {};
#pragma unroll
    for (int kk = 0; kk < DIMC; kk += 32) {
      const int ka = kk + lh * 8;        // A: interleaved K phases
      const int kb = kk + lh * 8;        // B: interleaved K phases
      V16U a, bmat;
      a.h[0]    = *(const v8h*)&sW1[row * W1S + ka];
      a.h[1]    = *(const v8h*)&sW1[row * W1S + ka + 16];
      bmat.h[0] = *(const v8h*)&sXn[col * XNS + kb];
      bmat.h[1] = *(const v8h*)&sXn[col * XNS + kb + 16];
      acc = __builtin_amdgcn_wmma_f32_16x16x32_f16(false, a.v, false, bmat.v,
                                                   (short)0, acc, false, false);
    }
    const int p  = nT * 16 + lq;
    if (p < NPIX) {
      const float vs = sValid[p];        // zero h outside image (conv zero-pads h)
      const int m0 = mT * 16 + lh * 8;   // D: vgpr r -> M = m0 + r
      v8h hv;
#pragma unroll
      for (int r = 0; r < 8; ++r)
        hv[r] = (_Float16)(gelu_exact(acc[r] + sB1[m0 + r]) * vs);
      *(v8h*)&sH[p * CHS + m0] = hv;
    }
  }
  __syncthreads();

  // ---------- Phase 3: stream dwconv+residual -> staging, then fc2 WMMA -> y ----------
  for (int g = 0; g < 4; ++g) {          // 128 interior pixels in 4 groups of 32
    // depthwise 3x3 conv + exact gelu + residual, 8 channels per task
#pragma unroll
    for (int it = 0; it < 4; ++it) {
      const int task = tid + it * 256;   // 32 px * 32 ch-groups = 1024 tasks
      const int pl   = task >> 5;        // local pixel 0..31
      const int ch0  = (task & 31) * 8;
      const int pi   = g * 32 + pl;
      const int py   = pi >> 4, px = pi & 15;
      float acc[8];
#pragma unroll
      for (int j = 0; j < 8; ++j) acc[j] = sBt[ch0 + j];
#pragma unroll
      for (int ky = 0; ky < 3; ++ky) {
#pragma unroll
        for (int kx = 0; kx < 3; ++kx) {
          const int hp = (py + ky) * HLW + (px + kx);   // halo offset -1 folded in
          const v8h hv = *(const v8h*)&sH[hp * CHS + ch0];
          const int tp = ky * 3 + kx;
#pragma unroll
          for (int j = 0; j < 8; ++j) acc[j] += sWt[tp * HIDC + ch0 + j] * (float)hv[j];
        }
      }
      const int hpc = (py + 1) * HLW + (px + 1);
      const v8h hc = *(const v8h*)&sH[hpc * CHS + ch0];
      v8h o;
#pragma unroll
      for (int j = 0; j < 8; ++j)
        o[j] = (_Float16)((float)hc[j] + gelu_exact(acc[j]));
      *(v8h*)&sStage[pl * CHS + ch0] = o;
    }
    __syncthreads();

    // fc2: Y[64 x 32] = W2[64x256] * Stage[256x32]; 4 M-tiles x 2 N-tiles = 8 waves
    {
      const int mT = wave & 3;
      const int nT = wave >> 2;
      const int row = mT * 16 + lq;
      const int col = nT * 16 + lq;
      v8f acc = {};
#pragma unroll
      for (int kk = 0; kk < HIDC; kk += 32) {
        const int ka = kk + lh * 8;
        const int kb = kk + lh * 8;
        V16U a, bmat;
        a.h[0]    = *(const v8h*)&sW2[row * CHS + ka];
        a.h[1]    = *(const v8h*)&sW2[row * CHS + ka + 16];
        bmat.h[0] = *(const v8h*)&sStage[col * CHS + kb];
        bmat.h[1] = *(const v8h*)&sStage[col * CHS + kb + 16];
        acc = __builtin_amdgcn_wmma_f32_16x16x32_f16(false, a.v, false, bmat.v,
                                                     (short)0, acc, false, false);
      }
      const int pl = nT * 16 + lq;
      const int pi = g * 32 + pl;
      const int py = pi >> 4, px = pi & 15;
      const int gy = tY * THh + py;
      const int gx = tX * TW  + px;
      const int m0 = mT * 16 + lh * 8;
      float* yp = y + (((size_t)bb * DIMC + m0) * IMGH + gy) * IMGW + gx;
#pragma unroll
      for (int r = 0; r < 8; ++r)
        yp[(size_t)r * (IMGH * IMGW)] = acc[r] + sB2[m0 + r];
    }
    __syncthreads();
  }
}

extern "C" void kernel_launch(void* const* d_in, const int* in_sizes, int n_in,
                              void* d_out, int out_size, void* d_ws, size_t ws_size,
                              hipStream_t stream) {
  const float* x     = (const float*)d_in[0];
  const float* ln_w  = (const float*)d_in[1];
  const float* ln_b  = (const float*)d_in[2];
  const float* fc1_w = (const float*)d_in[3];
  const float* fc1_b = (const float*)d_in[4];
  const float* pos_w = (const float*)d_in[5];
  const float* pos_b = (const float*)d_in[6];
  const float* fc2_w = (const float*)d_in[7];
  const float* fc2_b = (const float*)d_in[8];
  const float* s_sx  = (const float*)d_in[9];
  const float* b_sx  = (const float*)d_in[10];
  const float* s_sy  = (const float*)d_in[11];
  const float* b_sy  = (const float*)d_in[12];
  const float* s_lp  = (const float*)d_in[13];
  const float* b_lp  = (const float*)d_in[14];
  float* y = (float*)d_out;

  const int B = in_sizes[0] / (DIMC * IMGH * IMGW);
  dim3 grid(IMGW / TW, IMGH / THh, B);   // 8 x 16 x 16 = 2048 workgroups
  convnext_fused<<<grid, 256, 0, stream>>>(x, ln_w, ln_b, fc1_w, fc1_b, pos_w, pos_b,
                                           fc2_w, fc2_b, s_sx, b_sx, s_sy, b_sy,
                                           s_lp, b_lp, y);
}
